// RNNDecoder_6502580486600
// MI455X (gfx1250) — matrix operand order repaired and checked by reading, not compiled, not run on hardware
//
#include <hip/hip_runtime.h>

// ---------------------------------------------------------------------------
// RNN (GRU) decoder for MI455X / gfx1250.
// B=32, S=128, H=512, V=32000.  All GEMMs are NT (A[M,K] x W[N,K]^T), mapped
// onto v_wmma_f32_16x16x32_bf16 with fp32 accumulation.  The dominant output
// GEMM uses the Tensor Data Mover (tensor_load_to_lds) with double-buffered
// LDS tiles so DMA of chunk c+1 overlaps WMMA on chunk c.
// ---------------------------------------------------------------------------

#define BATCH 32
#define SLEN  128
#define HDIM  512
#define VDIM  32000

typedef __attribute__((ext_vector_type(16))) __bf16        v16bf;
typedef __attribute__((ext_vector_type(8)))  float         v8f;
typedef __attribute__((ext_vector_type(4)))  unsigned int  v4u;
typedef __attribute__((ext_vector_type(4)))  int           v4i;
typedef __attribute__((ext_vector_type(8)))  int           v8i;

#if defined(__has_builtin)
#if __has_builtin(__builtin_amdgcn_tensor_load_to_lds) && \
    __has_builtin(__builtin_amdgcn_s_wait_tensorcnt)
#define HAVE_TDM 1
#endif
#endif
#ifndef HAVE_TDM
#define HAVE_TDM 0
#endif

union Frag16 { uint4 u[2]; v16bf v; };   // 32 bytes = 16 bf16 = one WMMA A/B operand

__device__ __forceinline__ unsigned short f2bf(float f) {
  unsigned int u = __float_as_uint(f);
  u += 0x7FFFu + ((u >> 16) & 1u);       // round-to-nearest-even
  return (unsigned short)(u >> 16);
}

__device__ __forceinline__ v8f wmma_bf16(const Frag16& a, const Frag16& b, v8f c) {
  return __builtin_amdgcn_wmma_f32_16x16x32_bf16(false, a.v, false, b.v,
                                                 (short)0, c, false, false);
}

#if HAVE_TDM
// ---------------------------------------------------------------------------
// Issue one TDM 2-D tile load: tile_rows=128 rows x 32 bf16 K-elements from a
// [nrows, 512] bf16 row-major matrix into LDS, padding each 64B row with 16B
// (-> LDS row stride 40 ushorts = 80B).  D# bit layout per CDNA5 ISA ch.8.
// ---------------------------------------------------------------------------
__device__ __forceinline__ void tdm_load_tile(unsigned int lds_addr,
                                              const unsigned short* gptr,
                                              unsigned int nrows) {
  unsigned long long ga = (unsigned long long)(uintptr_t)gptr;
  v4u g0;
  g0[0] = 1u;                                            // count=1, user mode
  g0[1] = lds_addr;                                      // lds_addr [63:32]
  g0[2] = (unsigned int)(ga & 0xFFFFFFFFull);            // global_addr lo
  g0[3] = (unsigned int)((ga >> 32) & 0x01FFFFFFull)     // global_addr hi
          | (2u << 30);                                  // type=2 ("image")
  v8i g1;
  g1[0] = (int)((1u << 16)      // data_size = 1 -> 2-byte elements
              | (1u << 20)      // pad_enable
              | (3u << 22)      // pad_interval: 16 DWORDs (64B) per row
              | (3u << 25));    // pad_amount : 4 DWORDs (16B) pad
  g1[1] = (int)(512u << 16);                             // tensor_dim0 = 512
  g1[2] = (int)((nrows & 0xFFFFu) << 16);                // tensor_dim1 lo
  g1[3] = (int)(((nrows >> 16) & 0xFFFFu) | (32u << 16));// dim1 hi | tile_dim0=32
  g1[4] = (int)128u;                                     // tile_dim1=128, tile_dim2=0
  g1[5] = (int)512u;                                     // tensor_dim0_stride = 512
  g1[6] = 0;                                             // stride0 hi | stride1 lo
  g1[7] = 0;                                             // stride1 hi
  v4i z4 = {0, 0, 0, 0};
#if __clang_major__ >= 23
  v8i z8 = {0, 0, 0, 0, 0, 0, 0, 0};
  __builtin_amdgcn_tensor_load_to_lds(g0, g1, z4, z4, z8, 0);
#else
  __builtin_amdgcn_tensor_load_to_lds(g0, g1, z4, z4, 0);
#endif
}
#endif  // HAVE_TDM

// ---------------------------------------------------------------------------
// fp32 -> bf16 (bit) conversion, vectorized by 4.  n must be a multiple of 4.
// ---------------------------------------------------------------------------
__global__ void f32_to_bf16_kernel(const float* __restrict__ src,
                                   unsigned short* __restrict__ dst, int n) {
  int i = (blockIdx.x * 256 + threadIdx.x) * 4;
  if (i < n) {
    float4 d = *(const float4*)(src + i);
    dst[i + 0] = f2bf(d.x);
    dst[i + 1] = f2bf(d.y);
    dst[i + 2] = f2bf(d.z);
    dst[i + 3] = f2bf(d.w);
  }
}

// ---------------------------------------------------------------------------
// Embedding gather -> bf16 rows.  grid.x = B*S, block = 256.
// ---------------------------------------------------------------------------
__global__ void gather_embed_kernel(const int* __restrict__ tok,
                                    const float* __restrict__ emb,
                                    unsigned short* __restrict__ xbf) {
  int i = blockIdx.x;                 // row = b*S + s
  int t = threadIdx.x;
  const float* src = emb + (size_t)tok[i] * HDIM;
  unsigned short* dst = xbf + (size_t)i * HDIM;
  dst[t]       = f2bf(src[t]);
  dst[t + 256] = f2bf(src[t + 256]);
}

// ---------------------------------------------------------------------------
// Tiled NT GEMM:  C[M,N] = A[M,512](bf16) * Bm[N,512](bf16)^T + bias[N]
// WG tile 128x128, 256 threads = 8 waves (2x4), wave tile 64x32 (4x2 WMMA
// tiles).  K in chunks of 32, TDM double-buffered into padded LDS tiles
// (row stride 40 ushorts = 80B -> 20-dword bank spread, 16B aligned).
// ---------------------------------------------------------------------------
#define LDT2  40
#define TILEE (128 * LDT2)   // ushorts per tile buffer (5120 -> 10240B)

__global__ __launch_bounds__(256) void gemm_nt_bias_kernel(
    const unsigned short* __restrict__ A,    // [Mrows, 512] bf16
    const unsigned short* __restrict__ Bm,   // [N, 512] bf16
    const float* __restrict__ bias,          // [N]
    float* __restrict__ C,                   // [M, N] fp32
    int N, unsigned int Mrows) {
  __shared__ unsigned short tiles[4 * TILEE];   // A0 | A1 | B0 | B1 = 40960B

  const int t     = threadIdx.x;
  const int wave  = t >> 5;
  const int lane  = t & 31;
  const int lrow  = lane & 15;
  const int lhalf = lane >> 4;       // 0: lanes 0-15, 1: lanes 16-31
  const int m0    = blockIdx.y * 128;
  const int n0    = blockIdx.x * 128;
  const int waveM = wave & 1;        // 2 waves over M (64 rows each)
  const int waveN = wave >> 1;       // 4 waves over N (32 cols each)

  unsigned short* Abuf[2] = { tiles,             tiles + TILEE };
  unsigned short* Bbuf[2] = { tiles + 2 * TILEE, tiles + 3 * TILEE };

  v8f acc[4][2];
#pragma unroll
  for (int i = 0; i < 4; ++i)
#pragma unroll
    for (int j = 0; j < 2; ++j) acc[i][j] = (v8f){};

#if HAVE_TDM
  const unsigned int ldsbase = (unsigned int)(uintptr_t)tiles;
  // Prologue: DMA chunk 0 into buffer 0.
  if (t < 32) {
    tdm_load_tile(ldsbase + 0u * TILEE * 2u, A  + (size_t)m0 * HDIM, Mrows);
    tdm_load_tile(ldsbase + 2u * TILEE * 2u, Bm + (size_t)n0 * HDIM, N);
    __builtin_amdgcn_s_wait_tensorcnt(0);
  }
  __syncthreads();
#endif

  for (int c = 0; c < 16; ++c) {
    const int cur = c & 1;
#if HAVE_TDM
    // DMA next chunk into the other buffer while this chunk computes.
    if (t < 32 && c + 1 < 16) {
      const unsigned int nxt = (unsigned int)(cur ^ 1);
      tdm_load_tile(ldsbase + nxt * TILEE * 2u,
                    A + (size_t)m0 * HDIM + (c + 1) * 32, Mrows);
      tdm_load_tile(ldsbase + (2u + nxt) * TILEE * 2u,
                    Bm + (size_t)n0 * HDIM + (c + 1) * 32, N);
    }
#else
    // Fallback: synchronous VGPR staging of chunk c (512 uint4 per matrix).
#pragma unroll
    for (int i = 0; i < 2; ++i) {
      int f   = t + i * 256;         // 0..511 (uint4 index)
      int row = f >> 2;
      int seg = f & 3;
      *(uint4*)(&Abuf[cur][row * LDT2 + seg * 8]) =
          *(const uint4*)(A + (size_t)(m0 + row) * HDIM + c * 32 + seg * 8);
      *(uint4*)(&Bbuf[cur][row * LDT2 + seg * 8]) =
          *(const uint4*)(Bm + (size_t)(n0 + row) * HDIM + c * 32 + seg * 8);
    }
    __syncthreads();
#endif

    Frag16 a[4], b[2];
    // A fragment (16x32 bf16): lanes 0-15 K {0..7, 16..23},
    //                          lanes 16-31 K {8..15, 24..31}
#pragma unroll
    for (int sm = 0; sm < 4; ++sm) {
      const unsigned short* p =
          &Abuf[cur][(waveM * 64 + sm * 16 + lrow) * LDT2 + lhalf * 8];
      a[sm].u[0] = *(const uint4*)p;
      a[sm].u[1] = *(const uint4*)(p + 16);
    }
    // B fragment (32x16 bf16): lanes 0-15 K {0..7, 16..23}, lanes 16-31 K {8..15, 24..31}.
#pragma unroll
    for (int sn = 0; sn < 2; ++sn) {
      const unsigned short* p =
          &Bbuf[cur][(waveN * 32 + sn * 16 + lrow) * LDT2 + lhalf * 8];
      b[sn].u[0] = *(const uint4*)p;
      b[sn].u[1] = *(const uint4*)(p + 16);
    }
#pragma unroll
    for (int sm = 0; sm < 4; ++sm)
#pragma unroll
      for (int sn = 0; sn < 2; ++sn)
        acc[sm][sn] = wmma_bf16(a[sm], b[sn], acc[sm][sn]);

#if HAVE_TDM
    if (t < 32) __builtin_amdgcn_s_wait_tensorcnt(0);
#endif
    __syncthreads();
  }

  // Epilogue: C/D layout -> VGPR v holds row (v + 8*lhalf), col = lrow.
#pragma unroll
  for (int sn = 0; sn < 2; ++sn) {
    int n = n0 + waveN * 32 + sn * 16 + lrow;
    float bv = bias[n];
#pragma unroll
    for (int sm = 0; sm < 4; ++sm) {
      int mb = m0 + waveM * 64 + sm * 16 + 8 * lhalf;
#pragma unroll
      for (int v = 0; v < 8; ++v)
        C[(size_t)(mb + v) * N + n] = acc[sm][sn][v] + bv;
    }
  }
}

// ---------------------------------------------------------------------------
// One GRU step.  grid.x = 8 (H/64 column slices), block = 256 (8 waves).
// Each WG computes hg = h_prev @ W_hh^T for its three gate slices
// (rows {j0..j0+63, 512+j0.., 1024+j0..}) via WMMA, then fuses the
// sigmoid/tanh gate math and writes the fp32 carry + bf16 hidden.
// Dynamic LDS: hA (32x536 bf16, 34304B) | Bl (192x72 bf16, 27648B)
// with hg (32x200 fp32, 25600B) aliased over Bl after the K loop.
// ---------------------------------------------------------------------------
#define LDT 72
#define LA 536
#define SMEM_HA 34304
#define SMEM_TOTAL (34304 + 27648)

__global__ __launch_bounds__(256) void gru_step_kernel(
    const float* __restrict__ h_prev,        // [32, 512]
    float* __restrict__ h_next,              // [32, 512]
    const unsigned short* __restrict__ Whh,  // [1536, 512] bf16
    const float* __restrict__ xg,            // [4096, 1536] fp32
    const float* __restrict__ bhh,           // [1536]
    unsigned short* __restrict__ hseq,       // [4096, 512] bf16
    int s) {
  extern __shared__ char smem[];
  unsigned short* hA = (unsigned short*)smem;
  unsigned short* Bl = (unsigned short*)(smem + SMEM_HA);
  float*          hg = (float*)(smem + SMEM_HA);   // alias over Bl

  const int t     = threadIdx.x;
  const int wave  = t >> 5;
  const int lane  = t & 31;
  const int lrow  = lane & 15;
  const int lhalf = lane >> 4;
  const int j0    = blockIdx.x * 64;
  const int mt    = wave & 1;        // m-subtile (rows 0-15 / 16-31)
  const int npair = wave >> 1;       // 3 n-tiles each -> 12 n-tiles = 192 cols

  // Stage h_prev (fp32) as bf16 into LDS: full 32x512 A matrix.
#pragma unroll
  for (int i = 0; i < 16; ++i) {
    int f    = t + i * 256;          // float4 index, 0..4095
    int row  = f >> 7;
    int col4 = (f & 127) << 2;
    float4 d = *(const float4*)(h_prev + row * HDIM + col4);
    int base = row * LA + col4;
    hA[base + 0] = f2bf(d.x);
    hA[base + 1] = f2bf(d.y);
    hA[base + 2] = f2bf(d.z);
    hA[base + 3] = f2bf(d.w);
  }
  __syncthreads();

  v8f acc[3];
#pragma unroll
  for (int i = 0; i < 3; ++i) acc[i] = (v8f){};

  for (int c = 0; c < 8; ++c) {      // K chunks of 64
    // Stage 192 W_hh rows (gate-interleaved) x 64 K into LDS.
#pragma unroll
    for (int i = 0; i < 6; ++i) {
      int f   = t + i * 256;         // 0..1535 (uint4 index)
      int row = f >> 3;
      int seg = f & 7;
      int wr  = (row >> 6) * HDIM + j0 + (row & 63);
      *(uint4*)(&Bl[row * LDT + seg * 8]) =
          *(const uint4*)(Whh + (size_t)wr * HDIM + c * 64 + seg * 8);
    }
    __syncthreads();

#pragma unroll
    for (int ks = 0; ks < 2; ++ks) {
      const int kk = ks * 32;
      Frag16 a;
      {
        const unsigned short* p = &hA[(mt * 16 + lrow) * LA + c * 64 + kk + lhalf * 8];
        a.u[0] = *(const uint4*)p;
        a.u[1] = *(const uint4*)(p + 16);
      }
#pragma unroll
      for (int tt = 0; tt < 3; ++tt) {
        const unsigned short* p =
            &Bl[((npair * 3 + tt) * 16 + lrow) * LDT + kk + lhalf * 8];
        Frag16 b;
        b.u[0] = *(const uint4*)p;
        b.u[1] = *(const uint4*)(p + 16);
        acc[tt] = wmma_bf16(a, b, acc[tt]);
      }
    }
    __syncthreads();                 // Bl reusable next chunk / as hg after
  }

  // Dump accumulators (hg pre-activations) into LDS for cross-wave epilogue.
#pragma unroll
  for (int tt = 0; tt < 3; ++tt) {
    int ncol = (npair * 3 + tt) * 16 + lrow;     // 0..191 = gate*64 + col
#pragma unroll
    for (int v = 0; v < 8; ++v)
      hg[(mt * 16 + 8 * lhalf + v) * 200 + ncol] = acc[tt][v];
  }
  __syncthreads();

  // Fused GRU gate math for this WG's 32x64 slice.
#pragma unroll
  for (int i = 0; i < 8; ++i) {
    int idx = t + i * 256;           // 0..2047
    int m   = idx >> 6;
    int j   = idx & 63;
    int jj  = j0 + j;
    const float* xr = xg + (size_t)(m * SLEN + s) * (3 * HDIM);
    float hr = hg[m * 200 + j]         + bhh[jj];
    float hz = hg[m * 200 + 64 + j]    + bhh[HDIM + jj];
    float hn = hg[m * 200 + 128 + j]   + bhh[2 * HDIM + jj];
    float r  = 1.0f / (1.0f + __expf(-(xr[jj] + hr)));
    float z  = 1.0f / (1.0f + __expf(-(xr[HDIM + jj] + hz)));
    float nn = tanhf(xr[2 * HDIM + jj] + r * hn);
    float hp = h_prev[m * HDIM + jj];
    float hv = (1.0f - z) * nn + z * hp;
    h_next[m * HDIM + jj] = hv;
    hseq[(size_t)(m * SLEN + s) * HDIM + jj] = f2bf(hv);
  }
}

// ---------------------------------------------------------------------------
// Workspace layout (bytes, 256-aligned):
//   0        xbf    [4096,512]  bf16   4,194,304
//   4194304  WihB   [1536,512]  bf16   1,572,864
//   5767168  WhhB   [1536,512]  bf16   1,572,864
//   7340032  WoutB  [32000,512] bf16  32,768,000
//   40108032 xg     [4096,1536] f32   25,165,824
//   65273856 hseq   [4096,512]  bf16   4,194,304
//   69468160 h0     [32,512]    f32       65,536
//   69533696 h1     [32,512]    f32       65,536   (total ~69.6 MB)
// ---------------------------------------------------------------------------
extern "C" void kernel_launch(void* const* d_in, const int* in_sizes, int n_in,
                              void* d_out, int out_size, void* d_ws, size_t ws_size,
                              hipStream_t stream) {
  const int*   tok   = (const int*)  d_in[0];
  // d_in[1] = annotations (unused by reference)
  const float* hinit = (const float*)d_in[2];
  const float* emb   = (const float*)d_in[3];
  const float* Wih   = (const float*)d_in[4];
  const float* Whh   = (const float*)d_in[5];
  const float* bih   = (const float*)d_in[6];
  const float* bhh   = (const float*)d_in[7];
  const float* Wout  = (const float*)d_in[8];
  const float* bout  = (const float*)d_in[9];

  char* ws = (char*)d_ws;
  unsigned short* xbf   = (unsigned short*)(ws);
  unsigned short* WihB  = (unsigned short*)(ws + 4194304);
  unsigned short* WhhB  = (unsigned short*)(ws + 5767168);
  unsigned short* WoutB = (unsigned short*)(ws + 7340032);
  float*          xg    = (float*)         (ws + 40108032);
  unsigned short* hseq  = (unsigned short*)(ws + 65273856);
  float*          h0    = (float*)         (ws + 69468160);
  float*          h1    = (float*)         (ws + 69533696);

  // 1) One-time bf16 conversions (weights live in the 192MB L2 afterwards).
  f32_to_bf16_kernel<<<768,   256, 0, stream>>>(Wih,  WihB,  3 * HDIM * HDIM);
  f32_to_bf16_kernel<<<768,   256, 0, stream>>>(Whh,  WhhB,  3 * HDIM * HDIM);
  f32_to_bf16_kernel<<<16000, 256, 0, stream>>>(Wout, WoutB, VDIM * HDIM);
  gather_embed_kernel<<<BATCH * SLEN, 256, 0, stream>>>(tok, emb, xbf);
  hipMemcpyAsync(h0, hinit, BATCH * HDIM * sizeof(float),
                 hipMemcpyDeviceToDevice, stream);

  // 2) x_gates = xbf @ W_ih^T + b_ih   -> [4096, 1536] fp32
  gemm_nt_bias_kernel<<<dim3(12, 32), 256, 0, stream>>>(xbf, WihB, bih, xg,
                                                        3 * HDIM, BATCH * SLEN);

  // 3) Sequential GRU scan: 128 graph-captured step launches, ping-pong carry.
  float* hb[2] = {h0, h1};
  for (int s = 0; s < SLEN; ++s)
    gru_step_kernel<<<8, 256, SMEM_TOTAL, stream>>>(hb[s & 1], hb[(s + 1) & 1],
                                                    WhhB, xg, bhh, hseq, s);

  // 4) output = hiddens @ W_out^T + b_out -> [4096, 32000] fp32 (dominant GEMM)
  gemm_nt_bias_kernel<<<dim3(VDIM / 128, 32), 256, 0, stream>>>(
      hseq, WoutB, bout, (float*)d_out, VDIM, BATCH * SLEN);
}
